// CompiledBlock_45148696216108
// MI455X (gfx1250) — matrix operand order repaired and checked by reading, not compiled, not run on hardware
//
#include <hip/hip_runtime.h>

// ---------------------------------------------------------------------------
// out[j,:] = x[j,:] + LN(x[j,:]) @ W^T + b     (permute/inv-permute cancel:
// the block is row-wise, so gather followed by inverse scatter is identity).
//
// MI455X reasoning:
//   GEMM = 550 GFLOP; min HBM traffic ~1.5 GB => ~65 us @ 23.3 TB/s.
//   BF16 WMMA (K=32/instr) keeps compute at/below the bandwidth roof where
//   FP32 WMMA (K=4/instr) would be ~8x compute-bound. fp32 LN + fp32 accum.
//   Staging uses GLOBAL_LOAD_ASYNC_TO_LDS (ASYNCcnt) + double-buffered LDS;
//   epilogue X/out traffic is nontemporal to keep W + A tiles L2-resident.
// ---------------------------------------------------------------------------

#define N_TOK 65536
#define HID   2048
#define EPS_LN 1e-5f

typedef __attribute__((ext_vector_type(16))) __bf16 v16bf;
typedef __attribute__((ext_vector_type(8)))  float  v8f;

// Native clang vector (NOT HIP's struct int4) for the async-to-LDS builtin:
// param type per hipcc diagnostic is
//   '__attribute__((__vector_size__(4 * sizeof(int)))) int __device__ *'
typedef int v4i_vec __attribute__((vector_size(16)));
typedef __attribute__((address_space(1))) v4i_vec* as1_v4i;
typedef __attribute__((address_space(3))) v4i_vec* as3_v4i;

struct __align__(8) US4 { unsigned short x, y, z, w; };

static __device__ __forceinline__ unsigned short f2bf(float f) {
  union { float f; unsigned u; } c;
  c.f = f;
  unsigned u = c.u;
  unsigned r = u + 0x7fffu + ((u >> 16) & 1u);   // round to nearest even
  return (unsigned short)(r >> 16);
}

#if __has_builtin(__builtin_amdgcn_global_load_async_to_lds_b128)
#define USE_ASYNC_LDS 1
#else
#define USE_ASYNC_LDS 0
#endif

#if __has_builtin(__builtin_amdgcn_s_wait_asynccnt)
#define WAIT_ASYNC(n) __builtin_amdgcn_s_wait_asynccnt(n)
#else
#define WAIT_ASYNC(n) asm volatile("s_wait_asynccnt %0" ::"n"(n) : "memory")
#endif

// ---------------------------------------------------------------------------
// Kernel 1: per-row LayerNorm (fp32) -> bf16 activations in workspace.
// ---------------------------------------------------------------------------
__global__ void __launch_bounds__(256)
ln_to_bf16_kernel(const float* __restrict__ x,
                  const float* __restrict__ gamma,
                  const float* __restrict__ beta,
                  unsigned short* __restrict__ xn) {
  const int row = blockIdx.x;
  const int tid = threadIdx.x;
  const float* xr = x + (size_t)row * HID;

  float4 a0 = ((const float4*)xr)[tid];
  float4 a1 = ((const float4*)xr)[tid + 256];

  float s  = a0.x + a0.y + a0.z + a0.w + a1.x + a1.y + a1.z + a1.w;
  float sq = a0.x * a0.x + a0.y * a0.y + a0.z * a0.z + a0.w * a0.w +
             a1.x * a1.x + a1.y * a1.y + a1.z * a1.z + a1.w * a1.w;

  __shared__ float2 red[256];
  red[tid] = make_float2(s, sq);
  __syncthreads();
#pragma unroll
  for (int off = 128; off > 0; off >>= 1) {
    if (tid < off) {
      float2 o = red[tid + off];
      red[tid].x += o.x;
      red[tid].y += o.y;
    }
    __syncthreads();
  }
  const float mu  = red[0].x * (1.0f / HID);
  const float var = red[0].y * (1.0f / HID) - mu * mu;
  const float rs  = __frsqrt_rn(var + EPS_LN);

  float4 g0 = ((const float4*)gamma)[tid];
  float4 g1 = ((const float4*)gamma)[tid + 256];
  float4 b0 = ((const float4*)beta)[tid];
  float4 b1 = ((const float4*)beta)[tid + 256];

  US4 o0, o1;
  o0.x = f2bf((a0.x - mu) * rs * g0.x + b0.x);
  o0.y = f2bf((a0.y - mu) * rs * g0.y + b0.y);
  o0.z = f2bf((a0.z - mu) * rs * g0.z + b0.z);
  o0.w = f2bf((a0.w - mu) * rs * g0.w + b0.w);
  o1.x = f2bf((a1.x - mu) * rs * g1.x + b1.x);
  o1.y = f2bf((a1.y - mu) * rs * g1.y + b1.y);
  o1.z = f2bf((a1.z - mu) * rs * g1.z + b1.z);
  o1.w = f2bf((a1.w - mu) * rs * g1.w + b1.w);

  unsigned short* xnr = xn + (size_t)row * HID;
  ((US4*)xnr)[tid]       = o0;
  ((US4*)xnr)[tid + 256] = o1;
}

// ---------------------------------------------------------------------------
// Kernel 2: W (fp32) -> bf16 in workspace.
// ---------------------------------------------------------------------------
__global__ void __launch_bounds__(256)
w_to_bf16_kernel(const float* __restrict__ W, unsigned short* __restrict__ Wb) {
  const int i = blockIdx.x * 256 + threadIdx.x;
  float4 f = ((const float4*)W)[i];
  US4 o;
  o.x = f2bf(f.x);
  o.y = f2bf(f.y);
  o.z = f2bf(f.z);
  o.w = f2bf(f.w);
  ((US4*)Wb)[i] = o;
}

// ---------------------------------------------------------------------------
// Kernel 3: out = x + xn_bf16 @ Wb^T + bias  via v_wmma_f32_16x16x32_bf16.
// 128x128 block tile, BK=64, 8 waves x (2x4) 16x16 accumulators,
// double-buffered LDS fed by async-to-LDS loads (ASYNCcnt) when available.
// ---------------------------------------------------------------------------
#define BM 128
#define BN 128
#define BK 64
#define LSTR 80  // bf16 elems/row in LDS: 64 + 16 pad (160 B, 16B-aligned rows)

// Stage one 128x64 bf16 tile pair (A,B) into LDS buffers.
static __device__ __forceinline__ void stage_tiles(
    const unsigned short* __restrict__ A, const unsigned short* __restrict__ B,
    unsigned short* sA, unsigned short* sB, int bm0, int bn0, int k0, int tid) {
#if USE_ASYNC_LDS
#pragma unroll
  for (int p = 0; p < 4; ++p) {
    const int c = tid + p * 256;
    const int r = c >> 3;
    const int j = c & 7;
    __builtin_amdgcn_global_load_async_to_lds_b128(
        (as1_v4i)(A + (size_t)(bm0 + r) * HID + k0 + j * 8),
        (as3_v4i)(sA + r * LSTR + j * 8), 0, 0);
    __builtin_amdgcn_global_load_async_to_lds_b128(
        (as1_v4i)(B + (size_t)(bn0 + r) * HID + k0 + j * 8),
        (as3_v4i)(sB + r * LSTR + j * 8), 0, 0);
  }
#else
#pragma unroll
  for (int p = 0; p < 4; ++p) {
    const int c = tid + p * 256;
    const int r = c >> 3;
    const int j = c & 7;
    int4 va = *(const int4*)(A + (size_t)(bm0 + r) * HID + k0 + j * 8);
    *(int4*)(sA + r * LSTR + j * 8) = va;
    int4 vb = *(const int4*)(B + (size_t)(bn0 + r) * HID + k0 + j * 8);
    *(int4*)(sB + r * LSTR + j * 8) = vb;
  }
#endif
}

__global__ void __launch_bounds__(256)
gemm_bf16_wmma_kernel(const unsigned short* __restrict__ A,  // [N_TOK, HID] bf16
                      const unsigned short* __restrict__ B,  // [HID,  HID ] bf16 (W)
                      const float* __restrict__ X,           // residual (fp32)
                      const float* __restrict__ bias,        // [HID] fp32
                      float* __restrict__ out) {
  __shared__ __align__(16) unsigned short sA[2][BM * LSTR];  // 2 x 20 KB
  __shared__ __align__(16) unsigned short sB[2][BN * LSTR];  // 2 x 20 KB

  const int tid  = threadIdx.x;
  const int lane = tid & 31;
  const int wave = tid >> 5;   // 0..7
  const int wm   = wave & 3;   // 4 wave-rows (32 rows each)
  const int wn   = wave >> 2;  // 2 wave-cols (64 cols each)
  const int bn0  = blockIdx.x * BN;
  const int bm0  = blockIdx.y * BM;

  v8f acc[2][4];
#pragma unroll
  for (int i = 0; i < 2; ++i)
#pragma unroll
    for (int j = 0; j < 4; ++j) acc[i][j] = (v8f){0.f, 0.f, 0.f, 0.f, 0.f, 0.f, 0.f, 0.f};

  const int half = lane >> 4;  // 0: lanes 0-15, 1: lanes 16-31
  const int l15  = lane & 15;

  const int TITER = HID / BK;  // 32
  stage_tiles(A, B, sA[0], sB[0], bm0, bn0, 0, tid);

  for (int it = 0; it < TITER; ++it) {
    const int cur = it & 1;
    const bool has_next = (it + 1) < TITER;
    if (has_next)
      stage_tiles(A, B, sA[cur ^ 1], sB[cur ^ 1], bm0, bn0, (it + 1) * BK, tid);
#if USE_ASYNC_LDS
    // In-order completion: <=8 outstanding leaves only the prefetch in
    // flight, so the current tile's 8 async loads have landed in LDS.
    if (has_next) WAIT_ASYNC(8); else WAIT_ASYNC(0);
#endif
    __syncthreads();  // current tile visible to all waves

    const unsigned short* cA = sA[cur];
    const unsigned short* cB = sB[cur];
#pragma unroll
    for (int ks = 0; ks < 2; ++ks) {
      union Frag { int4 q[2]; v16bf v; };
      v16bf afrag[2], bfrag[4];

      // A fragment (16x32, 16-bit): half h: chunk0 = K[h*8..h*8+7] at byte
      // h*16, chunk1 = +32 bytes (K 16..23 / 24..31).
#pragma unroll
      for (int i = 0; i < 2; ++i) {
        const int row = wm * 32 + i * 16 + l15;
        const unsigned short* base = cA + row * LSTR + ks * 32 + half * 8;
        Frag u;
        u.q[0] = *(const int4*)(base);
        u.q[1] = *(const int4*)(base + 16);
        afrag[i] = u.v;
      }
      // B fragment (32x16): lane = column, half h: chunk0 = K[h*8..h*8+7],
      // chunk1 = +32 bytes (K 16..23 / 24..31), same as A.
#pragma unroll
      for (int j = 0; j < 4; ++j) {
        const int row = wn * 64 + j * 16 + l15;  // W row == B column
        const unsigned short* base = cB + row * LSTR + ks * 32 + half * 8;
        Frag u;
        u.q[0] = *(const int4*)(base);
        u.q[1] = *(const int4*)(base + 16);
        bfrag[j] = u.v;
      }
#pragma unroll
      for (int i = 0; i < 2; ++i)
#pragma unroll
        for (int j = 0; j < 4; ++j)
          acc[i][j] = __builtin_amdgcn_wmma_f32_16x16x32_bf16(
              false, afrag[i], false, bfrag[j], (short)0, acc[i][j], false, false);
    }
    __syncthreads();  // all waves done reading cur before it is re-staged
  }

  // --- epilogue: out = X + bias + acc (streaming, nontemporal) ------------
  // D layout: lanes 0-15: VGPR v -> (M=v, N=lane); lanes 16-31: (M=8+v, N=lane-16)
#pragma unroll
  for (int i = 0; i < 2; ++i) {
#pragma unroll
    for (int j = 0; j < 4; ++j) {
      const int gc      = bn0 + wn * 64 + j * 16 + l15;
      const int gr_base = bm0 + wm * 32 + i * 16 + half * 8;
      const float bj    = bias[gc];
#pragma unroll
      for (int v = 0; v < 8; ++v) {
        const size_t idx = (size_t)(gr_base + v) * HID + gc;
        const float xv = __builtin_nontemporal_load(&X[idx]);
        __builtin_nontemporal_store(xv + bj + acc[i][j][v], &out[idx]);
      }
    }
  }
}

// ---------------------------------------------------------------------------
// Host launcher. Inputs: x(0), permute(1, UNUSED - cancels), gamma(2),
// beta(3), W(4), b(5). Workspace: xn_bf16 (256 MB) + W_bf16 (8 MB).
// ---------------------------------------------------------------------------
extern "C" void kernel_launch(void* const* d_in, const int* in_sizes, int n_in,
                              void* d_out, int out_size, void* d_ws, size_t ws_size,
                              hipStream_t stream) {
  const float* x     = (const float*)d_in[0];
  const float* gamma = (const float*)d_in[2];
  const float* beta  = (const float*)d_in[3];
  const float* W     = (const float*)d_in[4];
  const float* b     = (const float*)d_in[5];
  float*       out   = (float*)d_out;

  unsigned short* xn = (unsigned short*)d_ws;
  unsigned short* Wb = (unsigned short*)((char*)d_ws + (size_t)N_TOK * HID * 2);

  ln_to_bf16_kernel<<<N_TOK, 256, 0, stream>>>(x, gamma, beta, xn);
  w_to_bf16_kernel<<<(HID * HID / 4) / 256, 256, 0, stream>>>(W, Wb);

  dim3 grid(HID / BN, N_TOK / BM);  // x = col-blocks (16), y = row-blocks (512)
  gemm_bf16_wmma_kernel<<<grid, 256, 0, stream>>>(xn, Wb, x, b, out);
}
